// sn_depthwise_cc_layerwise_44006234914838
// MI455X (gfx1250) — matrix operand order repaired and checked by reading, not compiled, not run on hardware
//
#include <hip/hip_runtime.h>
#include <hip/hip_bf16.h>
#include <math.h>

// ---------------------------------------------------------------------------
// CDNA5 WMMA / TDM types & helpers
// ---------------------------------------------------------------------------
typedef __attribute__((ext_vector_type(16))) __bf16        v16bf;
typedef __attribute__((ext_vector_type(8)))  float         v8f;
typedef __attribute__((ext_vector_type(4)))  unsigned int  v4u;
typedef __attribute__((ext_vector_type(8)))  int           v8i;
typedef __attribute__((ext_vector_type(4)))  int           v4i;

union Frag { v16bf v; uint4 q[2]; unsigned short u[16]; };

__device__ __forceinline__ unsigned short f2bf(float f) {
  unsigned int u = __float_as_uint(f);
  u += 0x7fffu + ((u >> 16) & 1u);          // RNE
  return (unsigned short)(u >> 16);
}
__device__ __forceinline__ float bf2f(unsigned short h) {
  return __uint_as_float((unsigned int)h << 16);
}

#if __has_builtin(__builtin_amdgcn_tensor_load_to_lds)
#define HAVE_TDM 1
#else
#define HAVE_TDM 0
#endif

#if HAVE_TDM
// Tensor Data Mover: DMA a 2-D tile (rows x dim0 bf16 elements, row stride in
// elements) from global memory into LDS. D# layout per CDNA5 ISA ch.8:
//  group0: [1:0]=count=1, [63:32]=lds_addr, [120:64]=global_addr, [127:126]=type=2
//  group1: [17:16]=data_size=1(2B), [79:48]=tensor_dim0, [111:80]=tensor_dim1,
//          [127:112]=tile_dim0, [143:128]=tile_dim1, [207:160]=tensor_dim0_stride
__device__ __forceinline__ void tdm_load_a(unsigned lds_off, const void* gaddr,
                                           unsigned dim0, unsigned rows, unsigned row_stride)
{
  unsigned long long ga = (unsigned long long)(size_t)gaddr;
  v4u g0 = { 1u,
             lds_off,
             (unsigned)(ga & 0xffffffffu),
             (unsigned)((ga >> 32) & 0x01ffffffu) | 0x80000000u };
  v8i g1 = { (int)(1u << 16),                                        // data_size = 2 bytes
             (int)((dim0 & 0xffffu) << 16),                          // tensor_dim0 lo
             (int)(((dim0 >> 16) & 0xffffu) | ((rows & 0xffffu) << 16)), // dim0 hi | tensor_dim1 lo
             (int)(((rows >> 16) & 0xffffu) | ((dim0 & 0xffffu) << 16)), // dim1 hi | tile_dim0
             (int)(rows & 0xffffu),                                  // tile_dim1 (tile_dim2 = 0)
             (int)row_stride,                                        // tensor_dim0_stride lo
             0, 0 };
  v4i g2 = {0, 0, 0, 0};
  v4i g3 = {0, 0, 0, 0};
#if defined(__clang_major__) && __clang_major__ >= 23
  v8i g4 = {0, 0, 0, 0, 0, 0, 0, 0};
  __builtin_amdgcn_tensor_load_to_lds(g0, g1, g2, g3, g4, 0);
#else
  __builtin_amdgcn_tensor_load_to_lds(g0, g1, g2, g3, 0);
#endif
}
#endif

// ---------------------------------------------------------------------------
// Fast implicit-GEMM conv, NHWC bf16 activations, weights bf16 [KH*KW][Cout][Cin].
// Requires Cin % 32 == 0 and Cout % 32 == 0 (all convs except the stem).
// Block = 128 threads (4 waves). Block tile M=32 (2 wmma acc/wave) x N=64.
// Per (kh,kw): one TDM tensor_load_to_lds DMAs the 32 x Cin bf16 A slab into LDS
// (wave 0 issues, s_wait_tensorcnt + barrier), then the Cin loop is pure:
//   4x ds_load_b128 (A frags) + 2x global_load_b128 (B, contiguous channels)
//   + 2x v_wmma_f32_16x16x32_bf16 per 32-wide K step.
// Epilogue: fused BN scale/shift + optional residual (bf16 NHWC) + ReLU.
// ---------------------------------------------------------------------------
__global__ __launch_bounds__(128)
void conv_nhwc_wmma(const unsigned short* __restrict__ x,
                    const unsigned short* __restrict__ w,
                    const float* __restrict__ scale, const float* __restrict__ shift,
                    const unsigned short* __restrict__ residual,
                    unsigned short* __restrict__ y,
                    int Cin, int H, int W, int Cout, int KH, int KW,
                    int stride, int pad, int Ho, int Wo, int relu)
{
  const int b    = blockIdx.z;
  const int m0   = blockIdx.y * 32;
  const int NT   = Ho * Wo;
  const int tid  = threadIdx.x;
  const int wave = tid >> 5;
  const int lane = tid & 31;
  const int hlf  = lane >> 4;      // fragment half (ISA 7.12.2 lane split)
  const int sub  = lane & 15;
  const int ncol = blockIdx.x * 64 + wave * 16 + sub;
  const bool n_ok = (ncol < NT);
  const int oh   = n_ok ? (ncol / Wo) : 0;
  const int ow   = n_ok ? (ncol - oh * Wo) : 0;
  const int ih0  = oh * stride - pad;
  const int iw0  = ow * stride - pad;

  __shared__ unsigned short As[32 * 256];     // A slab [32 m][Cin] bf16, up to 16 KB

  v8f acc0 = {0.f,0.f,0.f,0.f,0.f,0.f,0.f,0.f};
  v8f acc1 = {0.f,0.f,0.f,0.f,0.f,0.f,0.f,0.f};

  for (int kh = 0; kh < KH; ++kh) {
    const int ih = ih0 + kh;
    for (int kw = 0; kw < KW; ++kw) {
      const int iw = iw0 + kw;
      const bool b_ok = n_ok && ih >= 0 && ih < H && iw >= 0 && iw < W;
      const unsigned short* xb = x + ((size_t)(b * H + ih) * W + iw) * Cin;
      const unsigned short* wb = w + ((size_t)(kh * KW + kw) * Cout + m0) * Cin;

#if HAVE_TDM
      if (tid < 32) {                 // wave 0 drives the DMA (EXEC ignored by TDM)
        tdm_load_a((unsigned)(size_t)(void*)As, wb, (unsigned)Cin, 32u, (unsigned)Cin);
        __builtin_amdgcn_s_wait_tensorcnt(0);
      }
#else
      for (int i = tid * 8; i < 32 * Cin; i += 128 * 8)
        *(uint4*)(&As[i]) = *(const uint4*)(wb + i);
#endif
      __syncthreads();

      for (int c0 = 0; c0 < Cin; c0 += 32) {
        Frag a0, a1, bf;
        const int arow = sub * Cin + c0 + hlf * 8;
        a0.q[0] = *(const uint4*)(&As[arow]);
        a0.q[1] = *(const uint4*)(&As[arow + 16]);
        a1.q[0] = *(const uint4*)(&As[arow + 16 * Cin]);
        a1.q[1] = *(const uint4*)(&As[arow + 16 * Cin + 16]);

        if (b_ok) {
          bf.q[0] = *(const uint4*)(xb + c0 + hlf * 8);
          bf.q[1] = *(const uint4*)(xb + c0 + 16 + hlf * 8);
          __builtin_prefetch(xb + c0 + 32, 0, 0);   // global_prefetch_b8
        } else {
          bf.q[0] = make_uint4(0u, 0u, 0u, 0u);
          bf.q[1] = make_uint4(0u, 0u, 0u, 0u);
        }

        acc0 = __builtin_amdgcn_wmma_f32_16x16x32_bf16(false, a0.v, false, bf.v,
                                                       (short)0, acc0, false, false);
        acc1 = __builtin_amdgcn_wmma_f32_16x16x32_bf16(false, a1.v, false, bf.v,
                                                       (short)0, acc1, false, false);
      }
      __syncthreads();
    }
  }

  if (n_ok) {
    const size_t obase = ((size_t)(b * Ho + oh) * Wo + ow) * Cout;
#pragma unroll
    for (int i = 0; i < 8; ++i) {
      int ml = hlf * 8 + i;
      {
        int m = m0 + ml;
        float v = acc0[i] * scale[m] + shift[m];
        if (residual) v += bf2f(residual[obase + m]);
        if (relu) v = fmaxf(v, 0.0f);
        y[obase + m] = f2bf(v);
      }
      {
        int m = m0 + 16 + ml;
        float v = acc1[i] * scale[m] + shift[m];
        if (residual) v += bf2f(residual[obase + m]);
        if (relu) v = fmaxf(v, 0.0f);
        y[obase + m] = f2bf(v);
      }
    }
  }
}

// ---------------------------------------------------------------------------
// Stem conv: 7x7 s2 p3, Cin=1 (3 replicated channels folded), bn+relu fused.
// K = 49 padded to 64; weights bf16 [64][64] (zero-padded cols). M tile 16.
// ---------------------------------------------------------------------------
__global__ __launch_bounds__(128)
void conv1_wmma(const unsigned short* __restrict__ x,   // [B][299][299] bf16
                const unsigned short* __restrict__ w,   // [64][64] bf16
                const float* __restrict__ scale, const float* __restrict__ shift,
                unsigned short* __restrict__ y)          // [B][150][150][64] bf16
{
  const int H = 299, W = 299, Ho = 150, Wo = 150, NT = Ho * Wo;
  const int b = blockIdx.z, m0 = blockIdx.y * 16;
  const int tid = threadIdx.x, wave = tid >> 5, lane = tid & 31;
  const int hlf = lane >> 4, sub = lane & 15;
  const int ncol = blockIdx.x * 64 + wave * 16 + sub;
  const bool n_ok = (ncol < NT);
  const int oh = n_ok ? (ncol / Wo) : 0;
  const int ow = n_ok ? (ncol - oh * Wo) : 0;

  v8f acc = {0.f,0.f,0.f,0.f,0.f,0.f,0.f,0.f};
  const unsigned short* xb = x + (size_t)b * H * W;

  for (int k0 = 0; k0 < 64; k0 += 32) {
    Frag a, bf;
    const unsigned short* wr = w + (m0 + sub) * 64 + k0;
    a.q[0] = *(const uint4*)(wr + hlf * 8);
    a.q[1] = *(const uint4*)(wr + 16 + hlf * 8);
#pragma unroll
    for (int t = 0; t < 16; ++t) {
      int k = k0 + ((t < 8) ? (hlf * 8 + t) : (8 + hlf * 8 + (t - 8)));
      int kh = k / 7, kw = k - kh * 7;               // constant divisor -> mul/shift
      int ih = oh * 2 - 3 + kh, iw = ow * 2 - 3 + kw;
      unsigned short v = 0;
      if (n_ok && k < 49 && ih >= 0 && ih < H && iw >= 0 && iw < W)
        v = xb[ih * W + iw];
      bf.u[t] = v;
    }
    acc = __builtin_amdgcn_wmma_f32_16x16x32_bf16(false, a.v, false, bf.v,
                                                  (short)0, acc, false, false);
  }

  if (n_ok) {
    const size_t obase = ((size_t)(b * Ho + oh) * Wo + ow) * 64;
#pragma unroll
    for (int i = 0; i < 8; ++i) {
      int m = m0 + hlf * 8 + i;
      float v = fmaxf(acc[i] * scale[m] + shift[m], 0.0f);
      y[obase + m] = f2bf(v);
    }
  }
}

// ---------------------------------------------------------------------------
// Small prep / elementwise kernels (bf16 NHWC dataflow)
// ---------------------------------------------------------------------------
__global__ void bn_prep(const float* __restrict__ g, const float* __restrict__ bb,
                        const float* __restrict__ m, const float* __restrict__ v,
                        const float* __restrict__ bias,
                        float* __restrict__ scale, float* __restrict__ shift, int C)
{
  int i = blockIdx.x * blockDim.x + threadIdx.x;
  if (i < C) {
    float s  = g[i] * rsqrtf(v[i] + 1e-5f);
    float sh = bb[i] - m[i] * s;
    if (bias) sh += s * bias[i];
    scale[i] = s; shift[i] = sh;
  }
}

__global__ void fill_identity(float* scale, float* shift, int C) {
  int i = blockIdx.x * blockDim.x + threadIdx.x;
  if (i < C) { scale[i] = 1.0f; shift[i] = 0.0f; }
}

// fp32 [Cout][Cin][KH][KW] -> bf16 [KH*KW][Cout][Cin]
__global__ void wtransform(const float* __restrict__ src, unsigned short* __restrict__ dst,
                           int Cout, int Cin, int KHW)
{
  int idx = blockIdx.x * blockDim.x + threadIdx.x;
  int total = Cout * Cin * KHW;
  if (idx >= total) return;
  int c = idx % Cin;
  int r = idx / Cin;
  int o = r % Cout;
  int k = r / Cout;
  dst[idx] = f2bf(src[((size_t)o * Cin + c) * KHW + k]);
}

// fold 3 replicated input channels of conv1 -> bf16 [64][64] (49 valid, zero pad)
__global__ void conv1_fold_bf16(const float* __restrict__ w, unsigned short* __restrict__ weff)
{
  int i = blockIdx.x * blockDim.x + threadIdx.x;
  if (i < 64 * 64) {
    int o = i >> 6, k = i & 63;
    float v = 0.0f;
    if (k < 49) v = w[(o * 3 + 0) * 49 + k] + w[(o * 3 + 1) * 49 + k] + w[(o * 3 + 2) * 49 + k];
    weff[i] = f2bf(v);
  }
}

__global__ void cvt_f32_bf16(const float* __restrict__ x, unsigned short* __restrict__ y, int n)
{
  int i = blockIdx.x * blockDim.x + threadIdx.x;
  if (i < n) y[i] = f2bf(x[i]);
}

__global__ void maxpool_nhwc(const unsigned short* __restrict__ x, unsigned short* __restrict__ y,
                             int B, int H, int W, int C, int Ho, int Wo)
{
  int idx = blockIdx.x * blockDim.x + threadIdx.x;
  int total = B * Ho * Wo * C;
  if (idx >= total) return;
  int c = idx % C;
  int r = idx / C;
  int ox = r % Wo; r /= Wo;
  int oy = r % Ho;
  int b  = r / Ho;
  float mx = -INFINITY;
  for (int i = 0; i < 3; ++i) {
    int ih = oy * 2 - 1 + i; if (ih < 0 || ih >= H) continue;
    for (int j = 0; j < 3; ++j) {
      int iw = ox * 2 - 1 + j; if (iw < 0 || iw >= W) continue;
      mx = fmaxf(mx, bf2f(x[((size_t)(b * H + ih) * W + iw) * C + c]));
    }
  }
  y[idx] = f2bf(mx);
}

// depthwise cross-correlation, NHWC bf16 (channel contiguous -> coalesced)
__global__ void xcross_nhwc(const unsigned short* __restrict__ f1,
                            const unsigned short* __restrict__ f2,
                            unsigned short* __restrict__ out,
                            int B, int H, int W, int C, int pad, int Ho, int Wo)
{
  int idx = blockIdx.x * blockDim.x + threadIdx.x;
  int total = B * Ho * Wo * C;
  if (idx >= total) return;
  int c = idx % C;
  int r = idx / C;
  int ox = r % Wo; r /= Wo;
  int oy = r % Ho;
  int b  = r / Ho;
  int i0 = max(0, pad - oy), i1 = min(H, H + pad - oy);
  int j0 = max(0, pad - ox), j1 = min(W, W + pad - ox);
  const unsigned short* a = f1 + (size_t)b * H * W * C;
  const unsigned short* f = f2 + (size_t)b * H * W * C;
  float s = 0.0f;
  for (int i = i0; i < i1; ++i) {
    const unsigned short* ar = a + ((size_t)(oy - pad + i) * W + (ox - pad)) * C + c;
    const unsigned short* fr = f + (size_t)i * W * C + c;
    for (int j = j0; j < j1; ++j)
      s = fmaf(bf2f(ar[(size_t)j * C]), bf2f(fr[(size_t)j * C]), s);
  }
  out[idx] = f2bf(s);
}

__global__ void adaptive_pool_nhwc(const unsigned short* __restrict__ x,
                                   unsigned short* __restrict__ y,
                                   int B, int H, int W, int C, int OH, int OW)
{
  int idx = blockIdx.x * blockDim.x + threadIdx.x;
  int total = B * OH * OW * C;
  if (idx >= total) return;
  int c = idx % C;
  int r = idx / C;
  int ox = r % OW; r /= OW;
  int oy = r % OH;
  int b  = r / OH;
  int hs = (oy * H) / OH, he = ((oy + 1) * H + OH - 1) / OH;
  int ws = (ox * W) / OW, we = ((ox + 1) * W + OW - 1) / OW;
  const unsigned short* p = x + (size_t)b * H * W * C;
  float s = 0.0f;
  for (int i = hs; i < he; ++i)
    for (int j = ws; j < we; ++j)
      s += bf2f(p[((size_t)i * W + j) * C + c]);
  y[idx] = f2bf(s / (float)((he - hs) * (we - ws)));
}

// concat along channel + relu, NHWC: [P][128]+[P][128] -> [P][256]
__global__ void concat_relu_nhwc(const unsigned short* __restrict__ a,
                                 const unsigned short* __restrict__ b,
                                 unsigned short* __restrict__ y, int Pn, int C)
{
  int idx = blockIdx.x * blockDim.x + threadIdx.x;
  int total = Pn * 2 * C;
  if (idx >= total) return;
  int c = idx % (2 * C);
  int p = idx / (2 * C);
  float v = bf2f((c < C) ? a[(size_t)p * C + c] : b[(size_t)p * C + (c - C)]);
  y[idx] = f2bf(fmaxf(v, 0.0f));
}

// global average pool over HW + FC, x: NHWC bf16 [B][HW][C]
__global__ void gap_fc_nhwc(const unsigned short* __restrict__ x,
                            const float* __restrict__ fw, const float* __restrict__ fb,
                            float* __restrict__ out, int C, int HW, int nout)
{
  __shared__ float mean[256];
  int b = blockIdx.x, c = threadIdx.x;
  if (c < C) {
    float s = 0.0f;
    for (int i = 0; i < HW; ++i) s += bf2f(x[((size_t)b * HW + i) * C + c]);
    mean[c] = s / (float)HW;
  }
  __syncthreads();
  if (c < nout) {
    float s = fb[c];
    for (int k = 0; k < C; ++k) s = fmaf(fw[c * C + k], mean[k], s);
    out[b * nout + c] = s;
  }
}

// ---------------------------------------------------------------------------
// Host wiring. Flattened input order (jax-tree sorted keys):
//  0 img1, 1 img2, bb.bn1{b,g,m,v}=2..5, bb.conv1_w=6,
//  l1b0: bn1=7 bn2=11 w1=15 w2=16 ; l1b1: bn1=17 bn2=21 w1=25 w2=26
//  l2b0: bn1=27 bn2=31 dbn=35 dw=39 w1=40 w2=41 ; l2b1: bn1=42 bn2=46 w1=50 w2=51
//  l3b0: bn1=52 bn2=56 dbn=60 dw=64 w1=65 w2=66 ; l3b1: bn1=67 bn2=71 w1=75 w2=76
//  combos[0]: bias=77 bn=78 w=82 ; combos[1]: bias=83 bn=84 w=88
//  extra_w=89, fc_b=90, fc_w=91
// ---------------------------------------------------------------------------
static void run_conv_fast(hipStream_t s, const unsigned short* x, const unsigned short* w,
                          const float* const* P, int bn_base, const float* bias,
                          const unsigned short* residual, unsigned short* y,
                          float* scale, float* shift,
                          int B, int Cin, int H, int W, int Cout, int KH, int KW,
                          int stride, int pad, int relu)
{
  if (bn_base >= 0)
    bn_prep<<<(Cout + 255) / 256, 256, 0, s>>>(P[bn_base + 1], P[bn_base + 0],
                                               P[bn_base + 2], P[bn_base + 3],
                                               bias, scale, shift, Cout);
  else
    fill_identity<<<(Cout + 255) / 256, 256, 0, s>>>(scale, shift, Cout);

  int Ho = (H + 2 * pad - KH) / stride + 1;
  int Wo = (W + 2 * pad - KW) / stride + 1;
  int NT = Ho * Wo;
  dim3 grid((NT + 63) / 64, Cout / 32, B);
  conv_nhwc_wmma<<<grid, 128, 0, s>>>(x, w, scale, shift, residual, y,
                                      Cin, H, W, Cout, KH, KW, stride, pad, Ho, Wo, relu);
}

static void block_fast(hipStream_t s, const float* const* P,
                       const unsigned short* x, unsigned short* t1, unsigned short* t2,
                       unsigned short* out, float* scale, float* shift,
                       int B, int Cin, int H, int Cout, int stride,
                       int bn1, int bn2,
                       const unsigned short* w1, const unsigned short* w2,
                       const unsigned short* dw, int dbn)
{
  int Ho = (H - 1) / stride + 1;   // k=3, pad=1
  run_conv_fast(s, x, w1, P, bn1, nullptr, nullptr, t1, scale, shift,
                B, Cin, H, H, Cout, 3, 3, stride, 1, 1);
  const unsigned short* res = x;
  if (dw) {
    run_conv_fast(s, x, dw, P, dbn, nullptr, nullptr, t2, scale, shift,
                  B, Cin, H, H, Cout, 1, 1, stride, 0, 0);
    res = t2;
  }
  run_conv_fast(s, t1, w2, P, bn2, nullptr, res, out, scale, shift,
                B, Cout, Ho, Ho, Cout, 3, 3, 1, 1, 1);
}

extern "C" void kernel_launch(void* const* d_in, const int* in_sizes, int n_in,
                              void* d_out, int out_size, void* d_ws, size_t ws_size,
                              hipStream_t stream)
{
  (void)in_sizes; (void)n_in; (void)out_size; (void)ws_size;
  const float* const* P = (const float* const*)d_in;
  const int B = 16;

  // ---- carve workspace (byte allocator, 256B aligned chunks)
  char* base = (char*)d_ws;
  size_t off = 0;
  auto ab  = [&](size_t bytes) { char* p = base + off; off += (bytes + 255) & ~(size_t)255; return p; };
  auto af  = [&](size_t n) { return (float*)ab(n * sizeof(float)); };
  auto aus = [&](size_t n) { return (unsigned short*)ab(n * sizeof(unsigned short)); };

  float* scale = af(256);
  float* shift = af(256);

  // transformed weights (bf16)
  const int wt_idx[17] = {15,16,25,26, 39,40,41,50,51, 64,65,66,75,76, 82,88,89};
  const int wt_co [17] = {64,64,64,64, 128,128,128,128,128, 256,256,256,256,256, 256,256,128};
  const int wt_ci [17] = {64,64,64,64, 64, 64,128,128,128, 128,128,256,256,256, 256,256,256};
  const int wt_k  [17] = { 3, 3, 3, 3,  1,  3,  3,  3,  3,   1,  3,  3,  3,  3,   1,  1,  1};
  unsigned short* WT[17];
  for (int i = 0; i < 17; ++i) {
    int n = wt_co[i] * wt_ci[i] * wt_k[i] * wt_k[i];
    WT[i] = aus((size_t)n);
    wtransform<<<(n + 255) / 256, 256, 0, stream>>>(P[wt_idx[i]], WT[i],
                                                    wt_co[i], wt_ci[i], wt_k[i] * wt_k[i]);
  }
  unsigned short* weff = aus(64 * 64);
  conv1_fold_bf16<<<(64 * 64 + 255) / 256, 256, 0, stream>>>(P[6], weff);

  // activations (bf16, NHWC)
  unsigned short* imgb   = aus((size_t)B * 299 * 299);
  unsigned short* bufL   = aus((size_t)B * 150 * 150 * 64);
  unsigned short* buf1   = aus((size_t)B * 75 * 75 * 64);
  unsigned short* buf2   = aus((size_t)B * 75 * 75 * 64);
  unsigned short* buf3   = aus((size_t)B * 75 * 75 * 64);
  unsigned short* buf4   = aus((size_t)B * 75 * 75 * 64);
  unsigned short* fm38a  = aus((size_t)B * 38 * 38 * 128);
  unsigned short* fm38b  = aus((size_t)B * 38 * 38 * 128);
  unsigned short* fm19ea = aus((size_t)B * 19 * 19 * 128);
  unsigned short* fm19eb = aus((size_t)B * 19 * 19 * 128);
  unsigned short* sm38   = aus((size_t)B * 39 * 39 * 128);
  unsigned short* sm19   = aus((size_t)B * 19 * 19 * 128);
  unsigned short* sm38d  = aus((size_t)B * 19 * 19 * 128);
  unsigned short* cat    = aus((size_t)B * 19 * 19 * 256);
  unsigned short* cbuf   = aus((size_t)B * 19 * 19 * 256);

  // ---- backbone on both images
  for (int im = 0; im < 2; ++im) {
    unsigned short* fm38  = im ? fm38b  : fm38a;
    unsigned short* fm19e = im ? fm19eb : fm19ea;

    // img -> bf16
    {
      int n = B * 299 * 299;
      cvt_f32_bf16<<<(n + 255) / 256, 256, 0, stream>>>(P[im], imgb, n);
    }
    // stem: conv1 7x7 s2 p3 + bn1 + relu -> [B,150,150,64]
    bn_prep<<<1, 64, 0, stream>>>(P[3], P[2], P[4], P[5], nullptr, scale, shift, 64);
    {
      dim3 grid((150 * 150 + 63) / 64, 4, B);
      conv1_wmma<<<grid, 128, 0, stream>>>(imgb, weff, scale, shift, bufL);
    }
    // maxpool 3x3 s2 p1 -> [B,75,75,64]
    {
      int n = B * 75 * 75 * 64;
      maxpool_nhwc<<<(n + 255) / 256, 256, 0, stream>>>(bufL, buf1, B, 150, 150, 64, 75, 75);
    }
    // layer1
    block_fast(stream, P, buf1, buf2, buf4, buf3, scale, shift, B, 64, 75, 64, 1,
               7, 11, WT[0], WT[1], nullptr, 0);
    block_fast(stream, P, buf3, buf2, buf4, buf1, scale, shift, B, 64, 75, 64, 1,
               17, 21, WT[2], WT[3], nullptr, 0);
    // layer2 -> fm38 [B,38,38,128]
    block_fast(stream, P, buf1, buf2, buf3, buf4, scale, shift, B, 64, 75, 128, 2,
               27, 31, WT[5], WT[6], WT[4], 35);
    block_fast(stream, P, buf4, buf2, buf3, fm38, scale, shift, B, 128, 38, 128, 1,
               42, 46, WT[7], WT[8], nullptr, 0);
    // layer3 -> [B,19,19,256]
    block_fast(stream, P, fm38, buf2, buf3, buf4, scale, shift, B, 128, 38, 256, 2,
               52, 56, WT[10], WT[11], WT[9], 60);
    block_fast(stream, P, buf4, buf2, buf3, buf1, scale, shift, B, 256, 19, 256, 1,
               67, 71, WT[12], WT[13], nullptr, 0);
    // extra 1x1: 256 -> 128, no bn/relu
    run_conv_fast(stream, buf1, WT[16], P, -1, nullptr, nullptr, fm19e, scale, shift,
                  B, 256, 19, 19, 128, 1, 1, 1, 0, 0);
  }

  // ---- depthwise cross-correlations
  {
    int n = B * 39 * 39 * 128;
    xcross_nhwc<<<(n + 255) / 256, 256, 0, stream>>>(fm38a, fm38b, sm38, B, 38, 38, 128, 19, 39, 39);
  }
  {
    int n = B * 19 * 19 * 128;
    xcross_nhwc<<<(n + 255) / 256, 256, 0, stream>>>(fm19ea, fm19eb, sm19, B, 19, 19, 128, 9, 19, 19);
  }
  // ---- adaptive pool 39x39 -> 19x19
  {
    int n = B * 19 * 19 * 128;
    adaptive_pool_nhwc<<<(n + 255) / 256, 256, 0, stream>>>(sm38, sm38d, B, 39, 39, 128, 19, 19);
  }
  // ---- concat + relu -> [B,19,19,256]
  {
    int n = B * 361 * 256;
    concat_relu_nhwc<<<(n + 255) / 256, 256, 0, stream>>>(sm38d, sm19, cat, B * 361, 128);
  }
  // ---- combo blocks (1x1 conv, conv bias folded into bn shift, relu)
  run_conv_fast(stream, cat,  WT[14], P, 78, P[77], nullptr, cbuf, scale, shift,
                B, 256, 19, 19, 256, 1, 1, 1, 0, 1);
  run_conv_fast(stream, cbuf, WT[15], P, 84, P[83], nullptr, cat, scale, shift,
                B, 256, 19, 19, 256, 1, 1, 1, 0, 1);
  // ---- global average pool + FC -> [B,2]
  gap_fc_nhwc<<<B, 256, 0, stream>>>(cat, P[91], P[90], (float*)d_out, 256, 361, 2);
}
